// InformerEncoder_51178830299338
// MI455X (gfx1250) — matrix operand order repaired and checked by reading, not compiled, not run on hardware
//
#include <hip/hip_runtime.h>
#include <hip/hip_bf16.h>
#include <cmath>

typedef __bf16 bf16;
typedef __attribute__((ext_vector_type(16))) __bf16 v16bf;
typedef __attribute__((ext_vector_type(8)))  float  v8f;
typedef __attribute__((ext_vector_type(4)))  int    v4i_t;

static constexpr int BB   = 8;     // batch
static constexpr int L0   = 4096;  // initial sequence length
static constexpr int INF_ = 64;    // input feature dim
static constexpr int DM   = 512;   // model dim
static constexpr int DF   = 2048;  // ffn dim
static constexpr int NL   = 3;     // layers
static constexpr int HEADS= 8;
static constexpr int HD   = 64;    // head dim

#ifndef TRY_ASYNC
#define TRY_ASYNC 1
#endif
#if TRY_ASYNC && __has_builtin(__builtin_amdgcn_global_load_async_to_lds_b128)
#define USE_ASYNC_LDS 1
#else
#define USE_ASYNC_LDS 0
#endif

// ---------------------------------------------------------------- utilities
__device__ __forceinline__ unsigned hashu(unsigned x) {
    x ^= x >> 16; x *= 0x7feb352du; x ^= x >> 15; x *= 0x846ca68bu; x ^= x >> 16;
    return x;
}

__global__ void cvt_bf16_kernel(const float* __restrict__ in, bf16* __restrict__ out, size_t n) {
    size_t gid = (size_t)blockIdx.x * blockDim.x + threadIdx.x;
    if (gid < n) out[gid] = (bf16)in[gid];
}

// h[b,l,d] += pos_encoding(l,d)
__global__ void add_pos_kernel(float* __restrict__ h, int L, int D, size_t total) {
    size_t gid = (size_t)blockIdx.x * blockDim.x + threadIdx.x;
    if (gid >= total) return;
    int d = (int)(gid % D);
    int l = (int)((gid / D) % L);
    int i = d >> 1;
    float div = __expf(-(float)(2 * i) * (9.210340371976184f / (float)D)); // ln(10000)
    float ang = (float)l * div;
    h[gid] += (d & 1) ? __cosf(ang) : __sinf(ang);
}

// ---------------------------------------------------------------- WMMA GEMM
// C[M,N] = A[M,K](bf16, row-major) * W[N,K]^T(bf16, row-major) + bias, opt GELU
// tile 128x128, BK=32, 256 threads = 8 waves (2 across M x 4 across N),
// each wave: 4 (M) x 2 (N) = 8 WMMA accumulators. Double-buffered LDS staging;
// async global->LDS DMA (ASYNCcnt-tracked) on gfx1250.
__global__ __launch_bounds__(256)
void gemm_bf16_kernel(const bf16* __restrict__ A, const bf16* __restrict__ W,
                      const float* __restrict__ bias,
                      float* __restrict__ Cf, bf16* __restrict__ Cb,
                      int M, int N, int K, int gelu)
{
    __shared__ __align__(16) bf16 As[2][128 * 32];
    __shared__ __align__(16) bf16 Bs[2][128 * 32];
    const int tid  = threadIdx.x;
    const int lane = tid & 31;
    const int wave = tid >> 5;
    const int wm   = wave >> 2;      // 0..1 : 64 rows each
    const int wn   = wave & 3;       // 0..3 : 32 cols each
    const int bm   = blockIdx.x * 128;
    const int bn   = blockIdx.y * 128;

    v8f acc[4][2] = {};

    const int fr  = lane & 15;         // fragment row/col within 16
    const int akg = (lane >> 4) * 8;   // A: K base 0 or 8  (chunks at +0,+16)
    const int bkg = (lane >> 4) * 8;   // B: K base 0 or 8  (chunks at +0,+16)

    auto stage = [&](int kb, int buf) {
#pragma unroll
        for (int c = 0; c < 2; ++c) {
            int id  = (c << 8) + tid;           // 0..511
            int row = id >> 2;                  // 128 rows
            int off = (id & 3) * 8;             // 4 x 8 bf16 chunks
            const bf16* ga = &A[(size_t)(bm + row) * K + kb + off];
            const bf16* gb = &W[(size_t)(bn + row) * K + kb + off];
            bf16* la = &As[buf][row * 32 + off];
            bf16* lb = &Bs[buf][row * 32 + off];
#if USE_ASYNC_LDS
            {
                v4i_t* gav = (v4i_t*)const_cast<bf16*>(ga);
                v4i_t* gbv = (v4i_t*)const_cast<bf16*>(gb);
                v4i_t* lav = (v4i_t*)la;
                v4i_t* lbv = (v4i_t*)lb;
                __builtin_amdgcn_global_load_async_to_lds_b128(
                    (__attribute__((address_space(1))) v4i_t*)gav,
                    (__attribute__((address_space(3))) v4i_t*)lav, 0, 0);
                __builtin_amdgcn_global_load_async_to_lds_b128(
                    (__attribute__((address_space(1))) v4i_t*)gbv,
                    (__attribute__((address_space(3))) v4i_t*)lbv, 0, 0);
            }
#else
            *(uint4*)la = *(const uint4*)ga;
            *(uint4*)lb = *(const uint4*)gb;
#endif
            __builtin_prefetch(ga + 32, 0, 1);  // next K-step line
            __builtin_prefetch(gb + 32, 0, 1);
        }
    };

    auto stage_fence = [&]() {
#if USE_ASYNC_LDS
        asm volatile("s_wait_asynccnt 0x0" ::: "memory");
#endif
        __syncthreads();
    };

    auto compute = [&](int buf) {
        union Frag { v16bf v; uint4 q[2]; } fa[4], fb[2];
#pragma unroll
        for (int i = 0; i < 4; ++i) {
            const bf16* r = &As[buf][(wm * 64 + i * 16 + fr) * 32];
            fa[i].q[0] = *(const uint4*)&r[akg];
            fa[i].q[1] = *(const uint4*)&r[akg + 16];
        }
#pragma unroll
        for (int j = 0; j < 2; ++j) {
            const bf16* r = &Bs[buf][(wn * 32 + j * 16 + fr) * 32];
            fb[j].q[0] = *(const uint4*)&r[bkg];
            fb[j].q[1] = *(const uint4*)&r[bkg + 16];
        }
#pragma unroll
        for (int i = 0; i < 4; ++i)
#pragma unroll
            for (int j = 0; j < 2; ++j)
                acc[i][j] = __builtin_amdgcn_wmma_f32_16x16x32_bf16(
                    false, fa[i].v, false, fb[j].v, (short)0, acc[i][j], false, false);
    };

    const int steps = K / 32;
    stage(0, 0);
    stage_fence();
    for (int s = 0; s < steps; ++s) {
        int buf = s & 1;
        if (s + 1 < steps) stage((s + 1) * 32, buf ^ 1);
        compute(buf);
        stage_fence();
    }

    const int rhi = (lane >> 4) << 3;
#pragma unroll
    for (int i = 0; i < 4; ++i) {
        int gmB = bm + wm * 64 + i * 16 + rhi;
#pragma unroll
        for (int j = 0; j < 2; ++j) {
            int gn = bn + wn * 32 + j * 16 + fr;
            float bv = bias ? bias[gn] : 0.0f;
#pragma unroll
            for (int r = 0; r < 8; ++r) {
                float v = acc[i][j][r] + bv;
                if (gelu) v = 0.5f * v * (1.0f + erff(v * 0.70710678118f));
                size_t o = (size_t)(gmB + r) * N + gn;
                if (Cf) Cf[o] = v;
                if (Cb) Cb[o] = (bf16)v;
            }
        }
    }
}

// ---------------------------------------------------------------- attention
// M[l] = max_s qk_s - sum_s qk_s / L over u hashed-sampled keys
__global__ void attn_stats_kernel(const float* __restrict__ h, float* __restrict__ Mbuf,
                                  int L, int u, unsigned seed)
{
    int gid = blockIdx.x * blockDim.x + threadIdx.x;
    int total = BB * HEADS * L;
    if (gid >= total) return;
    int l = gid % L; int bh = gid / L; int head = bh % HEADS; int b = bh / HEADS;
    const float* base = h + ((size_t)b * L) * DM + head * HD;
    float4 q[16];
    const float4* qr = (const float4*)(base + (size_t)l * DM);
#pragma unroll
    for (int i = 0; i < 16; ++i) q[i] = qr[i];
    float mx = -1e30f, sm = 0.0f;
    for (int s = 0; s < u; ++s) {
        unsigned hs = hashu((unsigned)(l * 131u + s) * 2654435761u + seed * 0x9e3779b9u);
        int k = (int)(hs % (unsigned)L);
        const float4* kr = (const float4*)(base + (size_t)k * DM);
        float dot = 0.0f;
#pragma unroll
        for (int i = 0; i < 16; ++i) {
            float4 a = q[i], c = kr[i];
            dot += a.x * c.x + a.y * c.y + a.z * c.z + a.w * c.w;
        }
        mx = fmaxf(mx, dot); sm += dot;
    }
    Mbuf[gid] = mx - sm / (float)L;
}

// iterative top-u per (b,h); destroys Mbuf (scratch)
__global__ void topk_kernel(float* __restrict__ Mbuf, int* __restrict__ ibuf, int L, int u)
{
    int bh = blockIdx.x;
    float* Mr = Mbuf + (size_t)bh * L;
    __shared__ float mv[256];
    __shared__ int   mi[256];
    int tid = threadIdx.x;
    for (int slot = 0; slot < u; ++slot) {
        float best = -1e30f; int bi = 0;
        for (int l = tid; l < L; l += 256) {
            float v = Mr[l];
            if (v > best) { best = v; bi = l; }
        }
        mv[tid] = best; mi[tid] = bi;
        __syncthreads();
        for (int s = 128; s > 0; s >>= 1) {
            if (tid < s && mv[tid + s] > mv[tid]) { mv[tid] = mv[tid + s]; mi[tid] = mi[tid + s]; }
            __syncthreads();
        }
        if (tid == 0) { ibuf[bh * 64 + slot] = mi[0]; Mr[mi[0]] = -1e30f; }
        __syncthreads();
    }
}

// per-(b,head) column means of q over L
__global__ void colmean_kernel(const float* __restrict__ h, float* __restrict__ meanb, int L)
{
    int bh = blockIdx.x; int head = bh % HEADS; int b = bh / HEADS;
    int tid = threadIdx.x; int d = tid & 63, part = tid >> 6;
    __shared__ float red[256];
    const float* base = h + ((size_t)b * L) * DM + head * HD;
    int chunk = (L + 3) >> 2, k0 = part * chunk, k1 = min(L, k0 + chunk);
    float acc = 0.0f;
    for (int k = k0; k < k1; ++k) acc += base[(size_t)k * DM + d];
    red[tid] = acc;
    __syncthreads();
    if (part == 0)
        meanb[b * DM + head * HD + d] =
            (red[tid] + red[tid + 64] + red[tid + 128] + red[tid + 192]) / (float)L;
}

__global__ void fill_mean_kernel(const float* __restrict__ meanb, float* __restrict__ attn,
                                 int L, size_t total)
{
    size_t gid = (size_t)blockIdx.x * blockDim.x + threadIdx.x;
    if (gid >= total) return;
    int c = (int)(gid % DM);
    int b = (int)(gid / ((size_t)L * DM));
    attn[gid] = meanb[b * DM + c];
}

// full softmax attention for the top-u query rows; overwrites their ctx rows
__global__ void attn_topctx_kernel(const float* __restrict__ h, const int* __restrict__ ibuf,
                                   float* __restrict__ attn, int L, int u)
{
    int blk  = blockIdx.x;
    int slot = blk % u; int bh = blk / u; int head = bh % HEADS; int b = bh / HEADS;
    __shared__ float qred[HD];
    __shared__ float sc[4096];
    __shared__ float red[256];
    int tid = threadIdx.x;
    int l0  = ibuf[bh * 64 + slot];
    const float* base = h + ((size_t)b * L) * DM + head * HD;
    if (tid < HD) qred[tid] = base[(size_t)l0 * DM + tid];
    __syncthreads();

    float mx = -1e30f;
    for (int k = tid; k < L; k += 256) {
        const float* kr = base + (size_t)k * DM;
        float dot = 0.0f;
#pragma unroll
        for (int d = 0; d < HD; ++d) dot += qred[d] * kr[d];
        dot *= 0.125f; // 1/sqrt(64)
        sc[k] = dot;
        mx = fmaxf(mx, dot);
    }
    red[tid] = mx; __syncthreads();
    for (int s = 128; s > 0; s >>= 1) {
        if (tid < s) red[tid] = fmaxf(red[tid], red[tid + s]);
        __syncthreads();
    }
    mx = red[0]; __syncthreads();

    float sum = 0.0f;
    for (int k = tid; k < L; k += 256) { float e = __expf(sc[k] - mx); sc[k] = e; sum += e; }
    red[tid] = sum; __syncthreads();
    for (int s = 128; s > 0; s >>= 1) {
        if (tid < s) red[tid] += red[tid + s];
        __syncthreads();
    }
    float inv = 1.0f / red[0];
    __syncthreads();

    int d = tid & 63, part = tid >> 6;
    int chunk = (L + 3) >> 2, k0 = part * chunk, k1 = min(L, k0 + chunk);
    float acc = 0.0f;
    for (int k = k0; k < k1; ++k) acc += sc[k] * base[(size_t)k * DM + d];
    red[tid] = acc; __syncthreads();
    if (part == 0) {
        float c = (red[tid] + red[tid + 64] + red[tid + 128] + red[tid + 192]) * inv;
        attn[((size_t)b * L + l0) * DM + head * HD + d] = c;
    }
}

// ---------------------------------------------------------------- LN / distil
// out = LN(x + r) * s + b ; optional f32 + bf16 outputs ; D == 512
__global__ void add_ln_kernel(const float* __restrict__ x, const float* __restrict__ r,
                              const float* __restrict__ s, const float* __restrict__ bta,
                              float* __restrict__ outf, bf16* __restrict__ outb)
{
    const int D = DM;
    int row = blockIdx.x, tid = threadIdx.x;
    __shared__ float red[256];
    size_t base = (size_t)row * D;
    float v0 = x[base + tid]       + r[base + tid];
    float v1 = x[base + tid + 256] + r[base + tid + 256];
    red[tid] = v0 + v1;
    __syncthreads();
    for (int st = 128; st > 0; st >>= 1) { if (tid < st) red[tid] += red[tid + st]; __syncthreads(); }
    float mu = red[0] / (float)D;
    __syncthreads();
    float d0 = v0 - mu, d1 = v1 - mu;
    red[tid] = d0 * d0 + d1 * d1;
    __syncthreads();
    for (int st = 128; st > 0; st >>= 1) { if (tid < st) red[tid] += red[tid + st]; __syncthreads(); }
    float rs = rsqrtf(red[0] / (float)D + 1e-5f);
    float o0 = d0 * rs * s[tid]       + bta[tid];
    float o1 = d1 * rs * s[tid + 256] + bta[tid + 256];
    if (outf) { outf[base + tid] = o0; outf[base + tid + 256] = o1; }
    if (outb) { outb[base + tid] = (bf16)o0; outb[base + tid + 256] = (bf16)o1; }
}

__global__ void distil_kernel(const float* __restrict__ in, float* __restrict__ out,
                              int Lout, size_t total)
{
    size_t gid = (size_t)blockIdx.x * blockDim.x + threadIdx.x;
    if (gid >= total) return;
    int c = (int)(gid % DM);
    size_t row = gid / DM;
    int l2 = (int)(row % Lout);
    int b  = (int)(row / Lout);
    size_t ib = ((size_t)b * (Lout * 2) + 2 * l2) * DM + c;
    out[gid] = 0.5f * (in[ib] + in[ib + DM]);
}

// ---------------------------------------------------------------- host side
static inline size_t cdiv(size_t a, size_t b) { return (a + b - 1) / b; }

extern "C" void kernel_launch(void* const* d_in, const int* in_sizes, int n_in,
                              void* d_out, int out_size, void* d_ws, size_t ws_size,
                              hipStream_t stream)
{
    const float* x     = (const float*)d_in[0];
    const float* emb_w = (const float*)d_in[1];
    const float* emb_b = (const float*)d_in[2];
    const float* ln1_s = (const float*)d_in[3];
    const float* ln1_b = (const float*)d_in[4];
    const float* w1    = (const float*)d_in[5];
    const float* b1    = (const float*)d_in[6];
    const float* w2    = (const float*)d_in[7];
    const float* b2    = (const float*)d_in[8];
    const float* ln2_s = (const float*)d_in[9];
    const float* ln2_b = (const float*)d_in[10];
    const float* out_w = (const float*)d_in[11];
    const float* out_b = (const float*)d_in[12];
    float* out = (float*)d_out;

    // workspace carve
    char* p = (char*)d_ws;
    auto carve = [&](size_t bytes) { char* r = p; p += (bytes + 255) & ~(size_t)255; return (void*)r; };
    const size_t ACT = (size_t)BB * L0 * DM * sizeof(float);
    float* bufA    = (float*)carve(ACT);
    float* bufB    = (float*)carve(ACT);
    float* bufATTN = (float*)carve(ACT);
    float* bufXLN  = (float*)carve(ACT);
    bf16* xb    = (bf16*)carve((size_t)BB * L0 * INF_ * sizeof(bf16));
    bf16* xlnb  = (bf16*)carve((size_t)BB * L0 * DM   * sizeof(bf16));
    bf16* hidb  = (bf16*)carve((size_t)BB * L0 * DF   * sizeof(bf16));
    bf16* hfb   = (bf16*)carve((size_t)BB * 1024 * DM * sizeof(bf16));
    bf16* wbemb = (bf16*)carve((size_t)DM * INF_ * sizeof(bf16));
    bf16* wb1   = (bf16*)carve((size_t)NL * DF * DM * sizeof(bf16));
    bf16* wb2   = (bf16*)carve((size_t)NL * DM * DF * sizeof(bf16));
    bf16* wbout = (bf16*)carve((size_t)DM * DM * sizeof(bf16));
    float* Mbuf = (float*)carve((size_t)BB * HEADS * L0 * sizeof(float));
    int*  ibuf  = (int*)carve((size_t)BB * HEADS * 64 * sizeof(int));
    float* meanb= (float*)carve((size_t)BB * DM * sizeof(float));

    auto cvt = [&](const float* src, bf16* dst, size_t n) {
        cvt_bf16_kernel<<<dim3((unsigned)cdiv(n, 256)), 256, 0, stream>>>(src, dst, n);
    };
    // weight / input conversion to bf16
    cvt(x,     xb,    (size_t)BB * L0 * INF_);
    cvt(emb_w, wbemb, (size_t)DM * INF_);
    cvt(w1,    wb1,   (size_t)NL * DF * DM);
    cvt(w2,    wb2,   (size_t)NL * DM * DF);
    cvt(out_w, wbout, (size_t)DM * DM);

    auto gemm = [&](const bf16* A, const bf16* W, const float* bias,
                    float* Cf, bf16* Cb, int M, int N, int K, int gelu) {
        dim3 grid((unsigned)(M / 128), (unsigned)(N / 128));
        gemm_bf16_kernel<<<grid, 256, 0, stream>>>(A, W, bias, Cf, Cb, M, N, K, gelu);
    };

    // 1) embedding: h = x * emb_w^T + emb_b ; + positional encoding
    gemm(xb, wbemb, emb_b, bufA, nullptr, BB * L0, DM, INF_, 0);
    {
        size_t tot = (size_t)BB * L0 * DM;
        add_pos_kernel<<<dim3((unsigned)cdiv(tot, 256)), 256, 0, stream>>>(bufA, L0, DM, tot);
    }

    float* act = bufA;
    float* spare = bufB;
    int L = L0;
    for (int i = 0; i < NL; ++i) {
        int u = (int)fmin((double)L, fmax(1.0, 5.0 * ceil(log((double)L))));
        size_t totBLD = (size_t)BB * L * DM;

        // ProbSparse attention
        attn_stats_kernel<<<dim3((unsigned)cdiv((size_t)BB * HEADS * L, 256)), 256, 0, stream>>>(
            act, Mbuf, L, u, (unsigned)(i + 1));
        topk_kernel<<<dim3(BB * HEADS), 256, 0, stream>>>(Mbuf, ibuf, L, u);
        colmean_kernel<<<dim3(BB * HEADS), 256, 0, stream>>>(act, meanb, L);
        fill_mean_kernel<<<dim3((unsigned)cdiv(totBLD, 256)), 256, 0, stream>>>(meanb, bufATTN, L, totBLD);
        attn_topctx_kernel<<<dim3(BB * HEADS * u), 256, 0, stream>>>(act, ibuf, bufATTN, L, u);

        // LN1(x + attn) -> xln (f32 + bf16)
        add_ln_kernel<<<dim3(BB * L), 256, 0, stream>>>(act, bufATTN, ln1_s + i * DM, ln1_b + i * DM,
                                                        bufXLN, xlnb);
        // FFN: y = gelu(xln*W1^T+b1) (bf16) ; y2 = y*W2^T+b2 (f32, reuse bufATTN)
        gemm(xlnb, wb1 + (size_t)i * DF * DM, b1 + i * DF, nullptr, hidb, BB * L, DF, DM, 1);
        gemm(hidb, wb2 + (size_t)i * DM * DF, b2 + i * DM, bufATTN, nullptr, BB * L, DM, DF, 0);
        // LN2(xln + y2) -> act (and bf16 copy on last layer)
        add_ln_kernel<<<dim3(BB * L), 256, 0, stream>>>(bufXLN, bufATTN, ln2_s + i * DM, ln2_b + i * DM,
                                                        act, (i == NL - 1) ? hfb : nullptr);
        if (i < NL - 1) {
            int Lout = L / 2;
            size_t tot = (size_t)BB * Lout * DM;
            distil_kernel<<<dim3((unsigned)cdiv(tot, 256)), 256, 0, stream>>>(act, spare, Lout, tot);
            float* t = act; act = spare; spare = t;
            L = Lout;
        }
    }

    // output projection: out = h * out_w^T + out_b
    gemm(hfb, wbout, out_b, out, nullptr, BB * 1024, DM, DM, 0);
}
